// GATLayer_87162066305431
// MI455X (gfx1250) — matrix operand order repaired and checked by reading, not compiled, not run on hardware
//
#include <hip/hip_runtime.h>
#include <hip/hip_bf16.h>

// ---------------------------------------------------------------------------
// GAT layer, fused flash-style, tuned for MI455X (gfx1250, wave32, WMMA).
// B=8, N=2048, F_in=128, F_out=64.
//   Kernel 1: Wh = h@W  (fp32 WMMA 16x16x4), s1=Wh·a1, s2=Wh·a2,
//             Wh stored transposed as f16 [b][feat][node] for B-fragments.
//   Kernel 2: per 16-row tile, stream adj in 16x32 tiles (non-temporal: adj
//             is 134MB read-once; keep the 2MB WhT reuse set resident in the
//             192MB L2), masked leaky-relu scores, online softmax,
//             P(f16) x Wh(f16) via WMMA 16x16x32.
// ---------------------------------------------------------------------------

typedef __attribute__((ext_vector_type(16))) _Float16 v16h;
typedef __attribute__((ext_vector_type(8)))  _Float16 v8h;
typedef __attribute__((ext_vector_type(8)))  float    v8f;
typedef __attribute__((ext_vector_type(2)))  float    v2f;
typedef __attribute__((ext_vector_type(4)))  int      v4i;
typedef __attribute__((ext_vector_type(4)))  float    v4f;

#define BATCH  8
#define NNODE  2048
#define FIN    128
#define FOUT   64
#define NEG_BIG (-9e15f)
#define ALPHA  0.2f

// ---------------------------------------------------------------------------
// Kernel 1: projection + attention-vector reductions.
// One wave handles a 16-row tile: D(16x64) = A(16x128) x B(128x64), fp32 WMMA.
// A f32 16x4 layout: lanes 0-15 row=lane, K={k0,k0+1}; lanes 16-31 K={k0+2,k0+3}.
// B f32 4x16 layout (mirror): lanes 0-15 K={k0,k0+1}, N=lane; lanes 16-31 K+=2.
// ---------------------------------------------------------------------------
__global__ __launch_bounds__(256) void gat_proj_kernel(
    const float* __restrict__ h, const float* __restrict__ W,
    const float* __restrict__ a1, const float* __restrict__ a2,
    _Float16* __restrict__ WhT, float* __restrict__ s1, float* __restrict__ s2)
{
  const int lane = threadIdx.x & 31;
  const int wave = threadIdx.x >> 5;
  const int tile = blockIdx.x * 8 + wave;      // 1024 tiles total
  const int b    = tile >> 7;                  // 128 tiles per batch
  const int rowBase = (tile & 127) << 4;
  const int ln  = lane & 15;
  const int hi  = lane >> 4;                   // 0 or 1 (lane half)
  const int koff = hi * 2;

  v8f acc[4];
  #pragma unroll
  for (int t = 0; t < 4; ++t) acc[t] = (v8f)0.f;

  const float* hRow = h + ((size_t)(b * NNODE + rowBase + ln)) * FIN;

  #pragma unroll 4
  for (int k0 = 0; k0 < FIN; k0 += 4) {
    v2f afrag = *(const v2f*)(hRow + k0 + koff);
    #pragma unroll
    for (int t = 0; t < 4; ++t) {
      const int n = t * 16 + ln;
      v2f bfrag;
      bfrag.x = W[(k0 + koff) * FOUT + n];
      bfrag.y = W[(k0 + koff + 1) * FOUT + n];
      acc[t] = __builtin_amdgcn_wmma_f32_16x16x4_f32(
          false, afrag, false, bfrag, (short)0, acc[t], false, false);
    }
  }

  // --- s1 = Wh . a1 , s2 = Wh . a2 (row-wise dot over 64 features) ---------
  float a1v[4], a2v[4];
  #pragma unroll
  for (int t = 0; t < 4; ++t) { a1v[t] = a1[t * 16 + ln]; a2v[t] = a2[t * 16 + ln]; }

  float p1[8], p2[8];
  #pragma unroll
  for (int r = 0; r < 8; ++r) {
    float q1 = 0.f, q2 = 0.f;
    #pragma unroll
    for (int t = 0; t < 4; ++t) { q1 += acc[t][r] * a1v[t]; q2 += acc[t][r] * a2v[t]; }
    // reduce across the 16 lanes of this half-wave (each holds one feature col)
    #pragma unroll
    for (int off = 1; off < 16; off <<= 1) {
      q1 += __shfl_xor(q1, off, 32);
      q2 += __shfl_xor(q2, off, 32);
    }
    p1[r] = q1; p2[r] = q2;
  }

  // --- store WhT (f16, feature-major) + s1/s2 ------------------------------
  // C/D layout: element r of each v8f <-> row (r + 8*hi), N = 16*t + ln.
  #pragma unroll
  for (int t = 0; t < 4; ++t) {
    #pragma unroll
    for (int r = 0; r < 8; ++r) {
      const int rr = r + hi * 8;
      WhT[((size_t)(b * FOUT + t * 16 + ln)) * NNODE + rowBase + rr] =
          (_Float16)acc[t][r];
    }
  }
  if (ln == 0) {
    #pragma unroll
    for (int r = 0; r < 8; ++r) {
      s1[b * NNODE + rowBase + r + hi * 8] = p1[r];
      s2[b * NNODE + rowBase + r + hi * 8] = p2[r];
    }
  }
}

// ---------------------------------------------------------------------------
// Kernel 2: fused masked-softmax attention + aggregation.
// One wave per 16-row tile, K-loop over 32 columns at a time.
// f16 A 16x32 layout: lane<16 -> row=lane, K = {0..7} u {16..23};
//                     lane>=16 -> row=lane-16, K = {8..15} u {24..31}.
// f16 B 32x16 layout: lane half n: K = half*8 + {0..7} u 16+half*8 + {0..7}, N = lane&15 (two
//                     8-half runs from feature-major WhT, same K order as A -> 2x b128).
// adj loads use non-temporal hints (read-once 134MB stream).
// ---------------------------------------------------------------------------
__global__ __launch_bounds__(256) void gat_attn_kernel(
    const int* __restrict__ adj, const _Float16* __restrict__ WhT,
    const float* __restrict__ s1g, const float* __restrict__ s2g,
    float* __restrict__ out)
{
  const int lane = threadIdx.x & 31;
  const int wave = threadIdx.x >> 5;
  const int tile = blockIdx.x * 8 + wave;
  const int b    = tile >> 7;
  const int rowBase = (tile & 127) << 4;
  const int ln  = lane & 15;
  const int hi  = lane >> 4;
  const int kA  = hi * 8;                      // per-lane K base for A frag

  const int*      adjRow = adj + ((size_t)(b * NNODE + rowBase + ln)) * NNODE;
  const float*    s2b    = s2g + b * NNODE;
  const float     s1row  = s1g[b * NNODE + rowBase + ln];
  const _Float16* WhTb   = WhT + (size_t)b * FOUT * NNODE;

  v8f acc[4];
  #pragma unroll
  for (int t = 0; t < 4; ++t) acc[t] = (v8f)0.f;
  float m = -INFINITY;
  float l = 0.f;

  for (int j0 = 0; j0 < NNODE; j0 += 32) {
    // --- load 16x32 adj tile (non-temporal) + s2 slice in A-fragment order -
    v4i ad0 = __builtin_nontemporal_load((const v4i*)(adjRow + j0 + kA));
    v4i ad1 = __builtin_nontemporal_load((const v4i*)(adjRow + j0 + kA + 4));
    v4i ad2 = __builtin_nontemporal_load((const v4i*)(adjRow + j0 + kA + 16));
    v4i ad3 = __builtin_nontemporal_load((const v4i*)(adjRow + j0 + kA + 20));
    v4f sA = *(const v4f*)(s2b + j0 + kA);
    v4f sB = *(const v4f*)(s2b + j0 + kA + 4);
    v4f sC = *(const v4f*)(s2b + j0 + kA + 16);
    v4f sD = *(const v4f*)(s2b + j0 + kA + 20);

    auto score = [&](int a, float s2v) -> float {
      float x = s1row + s2v;
      float e = x > 0.f ? x : ALPHA * x;       // leaky relu
      return a > 0 ? e : NEG_BIG;              // adjacency mask
    };
    float sc[16];
    sc[0]  = score(ad0.x, sA.x); sc[1]  = score(ad0.y, sA.y);
    sc[2]  = score(ad0.z, sA.z); sc[3]  = score(ad0.w, sA.w);
    sc[4]  = score(ad1.x, sB.x); sc[5]  = score(ad1.y, sB.y);
    sc[6]  = score(ad1.z, sB.z); sc[7]  = score(ad1.w, sB.w);
    sc[8]  = score(ad2.x, sC.x); sc[9]  = score(ad2.y, sC.y);
    sc[10] = score(ad2.z, sC.z); sc[11] = score(ad2.w, sC.w);
    sc[12] = score(ad3.x, sD.x); sc[13] = score(ad3.y, sD.y);
    sc[14] = score(ad3.z, sD.z); sc[15] = score(ad3.w, sD.w);

    // --- online softmax (per-row stats live in lane pair {r, r+16}) --------
    float mx = sc[0];
    #pragma unroll
    for (int i = 1; i < 16; ++i) mx = fmaxf(mx, sc[i]);
    mx = fmaxf(mx, __shfl_xor(mx, 16, 32));    // join both K halves of row
    const float mnew = fmaxf(m, mx);
    const float fac  = __expf(m - mnew);

    float p[16], ps = 0.f;
    #pragma unroll
    for (int i = 0; i < 16; ++i) { p[i] = __expf(sc[i] - mnew); ps += p[i]; }
    ps += __shfl_xor(ps, 16, 32);
    l = l * fac + ps;
    m = mnew;

    // --- rescale accumulators: element r <-> row r + 8*hi ------------------
    #pragma unroll
    for (int r = 0; r < 8; ++r) {
      const float fr = __shfl(fac, r + kA, 32);
      #pragma unroll
      for (int t = 0; t < 4; ++t) acc[t][r] *= fr;
    }

    // --- P (f16) x Wh (f16) accumulate -------------------------------------
    v16h afrag;
    #pragma unroll
    for (int i = 0; i < 16; ++i) afrag[i] = (_Float16)p[i];

    #pragma unroll
    for (int t = 0; t < 4; ++t) {
      const _Float16* bp = WhTb + (size_t)(t * 16 + ln) * NNODE + j0 + hi * 8;
      v8h blo = *(const v8h*)(bp);
      v8h bhi = *(const v8h*)(bp + 16);
      v16h bfrag;
      #pragma unroll
      for (int i = 0; i < 8; ++i) { bfrag[i] = blo[i]; bfrag[8 + i] = bhi[i]; }
      acc[t] = __builtin_amdgcn_wmma_f32_16x16x32_f16(
          false, afrag, false, bfrag, (short)0, acc[t], false, false);
    }
  }

  // --- epilogue: normalize by running sum, ELU, non-temporal store ---------
  #pragma unroll
  for (int r = 0; r < 8; ++r) {
    const float lr = __shfl(l, r + kA, 32);
    const int rr = r + kA;                     // row index for element r
    const float inv = 1.f / lr;
    #pragma unroll
    for (int t = 0; t < 4; ++t) {
      float v = acc[t][r] * inv;
      float y = v > 0.f ? v : (__expf(v) - 1.f);   // ELU(alpha=1)
      __builtin_nontemporal_store(
          y, out + ((size_t)(b * NNODE + rowBase + rr)) * FOUT + t * 16 + ln);
    }
  }
}

// ---------------------------------------------------------------------------
extern "C" void kernel_launch(void* const* d_in, const int* in_sizes, int n_in,
                              void* d_out, int out_size, void* d_ws, size_t ws_size,
                              hipStream_t stream) {
  const float* h   = (const float*)d_in[0];
  const int*   adj = (const int*)d_in[1];
  const float* W   = (const float*)d_in[2];
  const float* a1  = (const float*)d_in[3];
  const float* a2  = (const float*)d_in[4];
  float* out = (float*)d_out;

  // workspace: WhT (f16, 2MB) | s1 | s2
  _Float16* WhT = (_Float16*)d_ws;
  float* s1 = (float*)((char*)d_ws + (size_t)BATCH * FOUT * NNODE * sizeof(_Float16));
  float* s2 = s1 + BATCH * NNODE;

  const int tiles  = BATCH * NNODE / 16;   // 1024 row-tiles
  const int blocks = tiles / 8;            // 8 waves (256 thr) per block

  gat_proj_kernel<<<blocks, 256, 0, stream>>>(h, W, a1, a2, WhT, s1, s2);
  gat_attn_kernel<<<blocks, 256, 0, stream>>>(adj, WhT, s1, s2, out);
}
